// ChannelBlock_54623394070602
// MI455X (gfx1250) — matrix operand order repaired and checked by reading, not compiled, not run on hardware
//
#include <hip/hip_runtime.h>
#include <hip/hip_bf16.h>
#include <math.h>
#include <stdint.h>

// ---------------------------------------------------------------------------
// ChannelBlock (XCiT-style) for MI455X / gfx1250, wave32, WMMA bf16,
// async Global->LDS double-buffered GEMM staging.
// ---------------------------------------------------------------------------

typedef __bf16 bf16_t;
typedef __attribute__((ext_vector_type(16))) __bf16 v16bf;
typedef __attribute__((ext_vector_type(8)))  __bf16 v8bf;
typedef __attribute__((ext_vector_type(8)))  float  v8f;
typedef __attribute__((ext_vector_type(4)))  int    v4i;

union FragAB { v16bf v; v8bf h[2]; bf16_t e[16]; };

static constexpr int BB   = 8;       // batch
static constexpr int NT   = 4096;    // tokens
static constexpr int C    = 512;     // dim
static constexpr int NH   = 8;       // heads
static constexpr int HD   = 64;      // head dim
static constexpr int HID  = 2048;    // mlp hidden
static constexpr int MROW = BB * NT; // 32768 rows
static constexpr float SCALE  = 0.125f;  // HD^-0.5
static constexpr float LN_EPS = 1e-5f;

// ----- CDNA5 async Global->LDS copy plumbing --------------------------------
#if defined(__has_builtin)
#if __has_builtin(__builtin_amdgcn_global_load_async_to_lds_b128)
#define HAVE_ASYNC_LDS 1
#endif
#endif

// Builtin prototype (from hipcc diagnostic): (v4i AS1*, v4i AS3*, int, int)
#define G128(p) ((__attribute__((address_space(1))) v4i*)(uintptr_t)(p))
#define L128(p) ((__attribute__((address_space(3))) v4i*)(unsigned)(uintptr_t)(p))

#if defined(__has_builtin)
#if __has_builtin(__builtin_amdgcn_s_wait_asynccnt)
#define WAIT_ASYNC(n) __builtin_amdgcn_s_wait_asynccnt(n)
#endif
#endif
#ifndef WAIT_ASYNC
#define WAIT_ASYNC(n) asm volatile("s_wait_asynccnt %0" ::"i"(n) : "memory")
#endif

__device__ __forceinline__ void copy16_g2l(const bf16_t* __restrict__ g,
                                           bf16_t* __restrict__ l) {
#if defined(HAVE_ASYNC_LDS)
  __builtin_amdgcn_global_load_async_to_lds_b128(G128(g), L128(l), 0, 0);
#else
  *(v8bf*)l = *(const v8bf*)g;
#endif
}

// ---------------------------------------------------------------------------
// Elementwise f32 -> bf16 cast (weights)
// ---------------------------------------------------------------------------
__global__ __launch_bounds__(256) void cast_f32_bf16(const float* __restrict__ in,
                                                     bf16_t* __restrict__ out, int n) {
  int i = blockIdx.x * blockDim.x + threadIdx.x;
  int stride = gridDim.x * blockDim.x;
  for (; i < n; i += stride) out[i] = (bf16_t)in[i];
}

// ---------------------------------------------------------------------------
// LayerNorm over last dim (512) -> bf16. One wave per row, 8 rows per block.
// ---------------------------------------------------------------------------
__global__ __launch_bounds__(256) void layernorm_bf16(const float* __restrict__ x,
                                                      const float* __restrict__ w,
                                                      const float* __restrict__ b,
                                                      bf16_t* __restrict__ out) {
  const int lane = threadIdx.x & 31;
  const int wid  = threadIdx.x >> 5;
  const int row  = blockIdx.x * 8 + wid;
  const float* xr = x + (size_t)row * C;

  float s = 0.f, sq = 0.f;
  float v[16];
#pragma unroll
  for (int j = 0; j < 16; ++j) {
    v[j] = xr[lane * 16 + j];
    s  += v[j];
    sq += v[j] * v[j];
  }
#pragma unroll
  for (int off = 16; off >= 1; off >>= 1) {
    s  += __shfl_xor(s,  off, 32);
    sq += __shfl_xor(sq, off, 32);
  }
  const float mu  = s * (1.0f / C);
  const float var = sq * (1.0f / C) - mu * mu;
  const float rs  = rsqrtf(var + LN_EPS);

  bf16_t* orow = out + (size_t)row * C;
#pragma unroll
  for (int j = 0; j < 16; ++j) {
    const int ch = lane * 16 + j;
    orow[ch] = (bf16_t)((v[j] - mu) * rs * w[ch] + b[ch]);
  }
}

// ---------------------------------------------------------------------------
// Generic bf16 GEMM:  out[M,N] = A[M,K] @ W[N,K]^T  (+bias)(+gelu)(+residual)
// Block = 256 threads (8 waves), 128x64 output tile, K-step 32.
// A/B tiles double-buffered in LDS via async Global->LDS b128 copies; rows
// padded to 40 halves (80 B) so the 16 fragment rows hit 16 distinct banks.
// Each wave computes a 32x32 tile: 4x v_wmma_f32_16x16x32_bf16 per K-step.
// ---------------------------------------------------------------------------
template <bool HAS_BIAS, bool GELU, bool HAS_RES, bool OUT_BF16>
__global__ __launch_bounds__(256) void gemm_bf16(const bf16_t* __restrict__ A,
                                                 const bf16_t* __restrict__ W,
                                                 const float* __restrict__ bias,
                                                 const float* __restrict__ res,
                                                 void* __restrict__ out,
                                                 int M, int N, int K) {
  constexpr int LDA = 40;  // padded row stride (halves) for 32-wide K tile
  __shared__ __align__(32) bf16_t lA[2][128 * LDA];
  __shared__ __align__(32) bf16_t lB[2][64 * LDA];

  const int tid  = threadIdx.x;
  const int lane = tid & 31;
  const int wid  = tid >> 5;
  const int m_blk = blockIdx.y * 128;
  const int n_blk = blockIdx.x * 64;
  const int m_wave = (wid >> 1) * 32;  // within block tile
  const int n_wave = (wid & 1) * 32;

  const int l16  = lane & 15;
  const int bsel = (lane >> 4) * 8;   // A-frag K sub-offset per half-wave
  const int kbB  = (lane >> 4) * 16;  // unused; B frag loads use bsel

  // Stage one 128x32 A tile + 64x32 B tile: 768 x 16B chunks, 3 per thread.
  auto stage = [&](int buf, int k0) {
#pragma unroll
    for (int i = 0; i < 2; ++i) {
      const int q   = i * 256 + tid;
      const int row = q >> 2;
      const int kc  = (q & 3) * 8;
      copy16_g2l(A + (size_t)(m_blk + row) * K + k0 + kc,
                 &lA[buf][row * LDA + kc]);
    }
    {
      const int row = tid >> 2;
      const int kc  = (tid & 3) * 8;
      copy16_g2l(W + (size_t)(n_blk + row) * K + k0 + kc,
                 &lB[buf][row * LDA + kc]);
    }
  };

  v8f acc[2][2];
#pragma unroll
  for (int i = 0; i < 2; ++i)
#pragma unroll
    for (int j = 0; j < 2; ++j) acc[i][j] = (v8f)(0.0f);

  const int nk = K >> 5;
  int buf = 0;
  stage(0, 0);

  for (int kt = 0; kt < nk; ++kt) {
    if (kt + 1 < nk) {
      stage(buf ^ 1, (kt + 1) * 32);
#if defined(HAVE_ASYNC_LDS)
      WAIT_ASYNC(3);  // the 3 just-issued copies may remain outstanding
#endif
    } else {
#if defined(HAVE_ASYNC_LDS)
      WAIT_ASYNC(0);
#endif
    }
    __syncthreads();

    FragAB af[2];
#pragma unroll
    for (int mf = 0; mf < 2; ++mf) {
      const bf16_t* p = &lA[buf][(m_wave + mf * 16 + l16) * LDA];
      af[mf].h[0] = *(const v8bf*)(p + bsel);
      af[mf].h[1] = *(const v8bf*)(p + 16 + bsel);
    }
    FragAB bfr[2];
#pragma unroll
    for (int nf = 0; nf < 2; ++nf) {
      const bf16_t* p = &lB[buf][(n_wave + nf * 16 + l16) * LDA];
      bfr[nf].h[0] = *(const v8bf*)(p + bsel);
      bfr[nf].h[1] = *(const v8bf*)(p + 16 + bsel);
    }
#pragma unroll
    for (int mf = 0; mf < 2; ++mf)
#pragma unroll
      for (int nf = 0; nf < 2; ++nf)
        acc[mf][nf] = __builtin_amdgcn_wmma_f32_16x16x32_bf16(
            false, af[mf].v, false, bfr[nf].v, (short)0, acc[mf][nf], false, false);

    __syncthreads();
    buf ^= 1;
  }

  // Epilogue: C/D layout -> lane l16 = col, (lane>>4)*8+r = row.
#pragma unroll
  for (int mf = 0; mf < 2; ++mf) {
#pragma unroll
    for (int nf = 0; nf < 2; ++nf) {
      const int col = n_blk + n_wave + nf * 16 + l16;
#pragma unroll
      for (int r = 0; r < 8; ++r) {
        const int row = m_blk + m_wave + mf * 16 + (lane >> 4) * 8 + r;
        float v = acc[mf][nf][r];
        if (HAS_BIAS) v += bias[col];
        if (GELU)     v = 0.5f * v * (1.0f + erff(v * 0.70710678118654752f));
        if (HAS_RES)  v += res[(size_t)row * N + col];
        if (OUT_BF16) ((bf16_t*)out)[(size_t)row * N + col] = (bf16_t)v;
        else          ((float*)out)[(size_t)row * N + col] = v;
      }
    }
  }
}

// ---------------------------------------------------------------------------
// Attention stage 1: per (b,h) compute P = softmax(SCALE * K^T V) over e.
// ---------------------------------------------------------------------------
__global__ __launch_bounds__(256) void attn_softmax(const bf16_t* __restrict__ qkv,
                                                    bf16_t* __restrict__ P) {
  const int bh = blockIdx.x;         // b*8 + h
  const int b  = bh >> 3;
  const int h  = bh & 7;
  const int lane = threadIdx.x & 31;
  const int wid  = threadIdx.x >> 5;
  const int dt   = wid >> 1;          // d-tile 0..3
  const int et0  = (wid & 1) * 2;     // first of two e-tiles

  __shared__ __align__(32) bf16_t lk[32][72];
  __shared__ __align__(32) bf16_t lv[32][72];
  __shared__ float S[64][66];

  v8f acc0 = (v8f)(0.0f), acc1 = (v8f)(0.0f);

  const int l16  = lane & 15;
  const int bsel = (lane >> 4) * 8;
  const int kbB  = (lane >> 4) * 16;
  const int ldr  = threadIdx.x >> 3;        // 0..31 tile row
  const int ldc  = (threadIdx.x & 7) * 8;   // 0..56 col chunk

  for (int n0 = 0; n0 < NT; n0 += 32) {
    const size_t base = ((size_t)(b * NT + n0 + ldr)) * (3 * C) + h * HD + ldc;
    *(v8bf*)&lk[ldr][ldc] = *(const v8bf*)(qkv + base + C);       // k
    *(v8bf*)&lv[ldr][ldc] = *(const v8bf*)(qkv + base + 2 * C);   // v
    __syncthreads();

    FragAB a;  // A = K^T tile: A[d, n_rel]
    const int d = dt * 16 + l16;
#pragma unroll
    for (int j = 0; j < 8; ++j) {
      a.e[j]     = lk[bsel + j][d];
      a.e[8 + j] = lk[16 + bsel + j][d];
    }
    FragAB b0, b1;  // B = V tile: B[n_rel, e]
#pragma unroll
    for (int j = 0; j < 16; ++j) {
      b0.e[j] = lv[(j < 8 ? bsel + j : 8 + bsel + j)][et0 * 16 + l16];
      b1.e[j] = lv[(j < 8 ? bsel + j : 8 + bsel + j)][(et0 + 1) * 16 + l16];
    }
    acc0 = __builtin_amdgcn_wmma_f32_16x16x32_bf16(false, a.v, false, b0.v,
                                                   (short)0, acc0, false, false);
    acc1 = __builtin_amdgcn_wmma_f32_16x16x32_bf16(false, a.v, false, b1.v,
                                                   (short)0, acc1, false, false);
    __syncthreads();
  }

#pragma unroll
  for (int r = 0; r < 8; ++r) {
    const int d = dt * 16 + (lane >> 4) * 8 + r;
    S[d][et0 * 16 + l16]       = acc0[r] * SCALE;
    S[d][(et0 + 1) * 16 + l16] = acc1[r] * SCALE;
  }
  __syncthreads();

  if (threadIdx.x < 64) {
    const int d = threadIdx.x;
    float m = -1e30f;
#pragma unroll 4
    for (int e = 0; e < 64; ++e) m = fmaxf(m, S[d][e]);
    float sum = 0.f;
#pragma unroll 4
    for (int e = 0; e < 64; ++e) { float t = expf(S[d][e] - m); S[d][e] = t; sum += t; }
    const float inv = 1.0f / sum;
    bf16_t* Pr = P + (size_t)bh * 4096 + d * 64;
#pragma unroll 4
    for (int e = 0; e < 64; ++e) Pr[e] = (bf16_t)(S[d][e] * inv);
  }
}

// ---------------------------------------------------------------------------
// Attention stage 2: out = Q @ P^T.  grid = (8 token-chunks, 64 bh).
// ---------------------------------------------------------------------------
__global__ __launch_bounds__(256) void attn_apply(const bf16_t* __restrict__ qkv,
                                                  const bf16_t* __restrict__ P,
                                                  bf16_t* __restrict__ attn_out) {
  const int chunk = blockIdx.x;  // 0..7 (512 tokens each)
  const int bh    = blockIdx.y;
  const int b     = bh >> 3;
  const int h     = bh & 7;
  const int lane  = threadIdx.x & 31;
  const int wid   = threadIdx.x >> 5;

  __shared__ __align__(32) bf16_t Pl[64 * 64];
  {
    const bf16_t* src = P + (size_t)bh * 4096;
    *(v8bf*)(Pl + threadIdx.x * 16)     = *(const v8bf*)(src + threadIdx.x * 16);
    *(v8bf*)(Pl + threadIdx.x * 16 + 8) = *(const v8bf*)(src + threadIdx.x * 16 + 8);
  }
  __syncthreads();

  const int l16  = lane & 15;
  const int bsel = (lane >> 4) * 8;
  const int kbB  = (lane >> 4) * 16;

  for (int it = 0; it < 4; ++it) {
    const int nf = chunk * 512 + (it * 8 + wid) * 16;  // 16-token frag base
    v8f acc[4];
#pragma unroll
    for (int dtile = 0; dtile < 4; ++dtile) acc[dtile] = (v8f)(0.0f);

#pragma unroll
    for (int e0 = 0; e0 < 64; e0 += 32) {
      const bf16_t* pA =
          qkv + (size_t)(b * NT + nf + l16) * (3 * C) + h * HD + e0;
      FragAB a;
      a.h[0] = *(const v8bf*)(pA + bsel);
      a.h[1] = *(const v8bf*)(pA + 16 + bsel);
#pragma unroll
      for (int dtile = 0; dtile < 4; ++dtile) {
        FragAB bf;
        const bf16_t* p = Pl + (dtile * 16 + l16) * 64 + e0;
        bf.h[0] = *(const v8bf*)(p + bsel);
        bf.h[1] = *(const v8bf*)(p + 16 + bsel);
        acc[dtile] = __builtin_amdgcn_wmma_f32_16x16x32_bf16(
            false, a.v, false, bf.v, (short)0, acc[dtile], false, false);
      }
    }
#pragma unroll
    for (int dtile = 0; dtile < 4; ++dtile) {
      const int d = dtile * 16 + l16;
#pragma unroll
      for (int r = 0; r < 8; ++r) {
        const int n = nf + (lane >> 4) * 8 + r;
        attn_out[(size_t)(b * NT + n) * C + h * HD + d] = (bf16_t)acc[dtile][r];
      }
    }
  }
}

// ---------------------------------------------------------------------------
// ECA pooled mean: pooled[b,c] = mean of contiguous 4096-float chunk of y.
// ---------------------------------------------------------------------------
__global__ __launch_bounds__(256) void eca_pool(const float* __restrict__ y,
                                                float* __restrict__ pooled) {
  const int bc = blockIdx.x;  // b*512 + c
  const float* p = y + (size_t)bc * 4096;
  float s = 0.f;
  for (int i = threadIdx.x; i < 4096; i += 256) s += p[i];
  __shared__ float red[256];
  red[threadIdx.x] = s;
  __syncthreads();
  for (int off = 128; off >= 1; off >>= 1) {
    if (threadIdx.x < off) red[threadIdx.x] += red[threadIdx.x + off];
    __syncthreads();
  }
  if (threadIdx.x == 0) pooled[bc] = red[0] * (1.0f / 4096.0f);
}

__global__ __launch_bounds__(256) void eca_gate(const float* __restrict__ pooled,
                                                const float* __restrict__ w,
                                                float* __restrict__ gate) {
  const int i = blockIdx.x * blockDim.x + threadIdx.x;  // 0..4095
  if (i >= BB * C) return;
  const int c = i & (C - 1);
  const float left  = (c > 0)     ? pooled[i - 1] : 0.f;
  const float mid   = pooled[i];
  const float right = (c < C - 1) ? pooled[i + 1] : 0.f;
  const float conv  = w[0] * left + w[1] * mid + w[2] * right;
  gate[i] = 1.0f / (1.0f + expf(-conv));
}

// ---------------------------------------------------------------------------
// Final: out[b,s,c] = x2[b,s,c] + y_flat[b][c*4096 + s] * (1 + gate[b,c])
// ---------------------------------------------------------------------------
__global__ __launch_bounds__(256) void final_combine(const float* __restrict__ x2,
                                                     const float* __restrict__ y,
                                                     const float* __restrict__ gate,
                                                     float* __restrict__ out) {
  __shared__ float tile[32][33];
  const int s0 = blockIdx.x * 32;
  const int c0 = blockIdx.y * 32;
  const int b  = blockIdx.z;
  const int tx = threadIdx.x;   // 0..31
  const int ty = threadIdx.y;   // 0..7

#pragma unroll
  for (int i = 0; i < 4; ++i) {
    const int c = c0 + ty + i * 8;
    tile[ty + i * 8][tx] = y[(size_t)b * (C * NT) + (size_t)c * NT + (s0 + tx)];
  }
  __syncthreads();

  const float g = 1.0f + gate[b * C + c0 + tx];
#pragma unroll
  for (int i = 0; i < 4; ++i) {
    const int s = s0 + ty + i * 8;
    const size_t idx = ((size_t)b * NT + s) * C + (c0 + tx);
    out[idx] = x2[idx] + tile[tx][ty + i * 8] * g;
  }
}

// ---------------------------------------------------------------------------
// Launcher
// ---------------------------------------------------------------------------
static inline size_t align256(size_t x) { return (x + 255) & ~(size_t)255; }

extern "C" void kernel_launch(void* const* d_in, const int* in_sizes, int n_in,
                              void* d_out, int out_size, void* d_ws, size_t ws_size,
                              hipStream_t stream) {
  const float* x      = (const float*)d_in[0];
  const float* ln1_w  = (const float*)d_in[1];
  const float* ln1_b  = (const float*)d_in[2];
  const float* qkv_w  = (const float*)d_in[3];
  const float* proj_w = (const float*)d_in[4];
  const float* proj_b = (const float*)d_in[5];
  const float* ln2_w  = (const float*)d_in[6];
  const float* ln2_b  = (const float*)d_in[7];
  const float* fc1_w  = (const float*)d_in[8];
  const float* fc1_b  = (const float*)d_in[9];
  const float* fc2_w  = (const float*)d_in[10];
  const float* fc2_b  = (const float*)d_in[11];
  const float* eca_w  = (const float*)d_in[12];
  float* out = (float*)d_out;

  char* ws = (char*)d_ws;
  size_t o = 0;
  bf16_t* w_qkv  = (bf16_t*)(ws + o); o = align256(o + (size_t)3 * C * C * 2);
  bf16_t* w_proj = (bf16_t*)(ws + o); o = align256(o + (size_t)C * C * 2);
  bf16_t* w_fc1  = (bf16_t*)(ws + o); o = align256(o + (size_t)HID * C * 2);
  bf16_t* w_fc2  = (bf16_t*)(ws + o); o = align256(o + (size_t)C * HID * 2);
  bf16_t* xln    = (bf16_t*)(ws + o); o = align256(o + (size_t)MROW * C * 2);
  bf16_t* qkv    = (bf16_t*)(ws + o);             // union {qkv, h}
  bf16_t* hbuf   = qkv;
  o = align256(o + (size_t)MROW * HID * 2);       // max(3C, HID) = HID
  bf16_t* attn_o = (bf16_t*)(ws + o); o = align256(o + (size_t)MROW * C * 2);
  bf16_t* Pmat   = (bf16_t*)(ws + o); o = align256(o + (size_t)BB * NH * HD * HD * 2);
  float*  x2     = (float*)(ws + o);  o = align256(o + (size_t)MROW * C * 4);
  float*  ybuf   = (float*)(ws + o);  o = align256(o + (size_t)MROW * C * 4);
  float*  pooled = (float*)(ws + o);  o = align256(o + (size_t)BB * C * 4);
  float*  gate   = (float*)(ws + o);  o = align256(o + (size_t)BB * C * 4);
  (void)ws_size; (void)n_in; (void)in_sizes; (void)out_size;

  cast_f32_bf16<<<1024, 256, 0, stream>>>(qkv_w,  w_qkv,  3 * C * C);
  cast_f32_bf16<<<512,  256, 0, stream>>>(proj_w, w_proj, C * C);
  cast_f32_bf16<<<1024, 256, 0, stream>>>(fc1_w,  w_fc1,  HID * C);
  cast_f32_bf16<<<1024, 256, 0, stream>>>(fc2_w,  w_fc2,  C * HID);

  layernorm_bf16<<<MROW / 8, 256, 0, stream>>>(x, ln1_w, ln1_b, xln);

  gemm_bf16<false, false, false, true>
      <<<dim3((3 * C) / 64, MROW / 128), 256, 0, stream>>>(
          xln, w_qkv, nullptr, nullptr, qkv, MROW, 3 * C, C);

  attn_softmax<<<BB * NH, 256, 0, stream>>>(qkv, Pmat);
  attn_apply<<<dim3(8, BB * NH), 256, 0, stream>>>(qkv, Pmat, attn_o);

  gemm_bf16<true, false, true, false>
      <<<dim3(C / 64, MROW / 128), 256, 0, stream>>>(
          attn_o, w_proj, proj_b, x, x2, MROW, C, C);

  layernorm_bf16<<<MROW / 8, 256, 0, stream>>>(x2, ln2_w, ln2_b, xln);

  gemm_bf16<true, true, false, true>
      <<<dim3(HID / 64, MROW / 128), 256, 0, stream>>>(
          xln, w_fc1, fc1_b, nullptr, hbuf, MROW, HID, C);

  gemm_bf16<true, false, false, false>
      <<<dim3(C / 64, MROW / 128), 256, 0, stream>>>(
          hbuf, w_fc2, fc2_b, nullptr, ybuf, MROW, C, HID);

  eca_pool<<<BB * C, 256, 0, stream>>>(ybuf, pooled);
  eca_gate<<<(BB * C + 255) / 256, 256, 0, stream>>>(pooled, eca_w, gate);

  final_combine<<<dim3(NT / 32, C / 32, BB), dim3(32, 8), 0, stream>>>(
      x2, ybuf, gate, out);
}
